// GAT_46909632807647
// MI455X (gfx1250) — matrix operand order repaired and checked by reading, not compiled, not run on hardware
//
#include <hip/hip_runtime.h>
#include <hip/hip_bf16.h>
#include <math.h>

// ---------------------------------------------------------------------------
// Types for CDNA5 WMMA (wave32)
// ---------------------------------------------------------------------------
typedef __attribute__((ext_vector_type(16))) __bf16 bf16x16;
typedef __attribute__((ext_vector_type(8)))  __bf16 bf16x8;
typedef __attribute__((ext_vector_type(8)))  float  f32x8;

static constexpr int GN   = 20000;   // nodes
static constexpr int GE   = 80000;   // edges (before self loops)
static constexpr int GNG  = 64;      // graphs
static constexpr int GEMB = 780;     // embedding size
static constexpr int GMP  = 20096;   // ceil(GN/128)*128  (padded rows)

__device__ __forceinline__ __bf16 to_bf16(float f) {
    unsigned u = __float_as_uint(f);
    unsigned r = (u + 0x7FFFu + ((u >> 16) & 1u)) >> 16;   // RNE truncate
    unsigned short s = (unsigned short)r;
    return __builtin_bit_cast(__bf16, s);
}

__device__ __forceinline__ unsigned pack2(float a, float b) {
    return (unsigned)__builtin_bit_cast(unsigned short, to_bf16(a)) |
           ((unsigned)__builtin_bit_cast(unsigned short, to_bf16(b)) << 16);
}

// Low 32 bits of a generic LDS pointer == LDS byte offset (aperture in high bits)
__device__ __forceinline__ unsigned lds_u32(const void* p) {
    return (unsigned)(uintptr_t)p;
}

// 32-byte global->LDS async copy (2x b128). INST_OFFSET applies to both the
// memory and LDS addresses, so one address pair covers both halves.
__device__ __forceinline__ void async_copy_32B(unsigned lds, const void* g) {
    asm volatile(
        "global_load_async_to_lds_b128 %0, %1, off\n\t"
        "global_load_async_to_lds_b128 %0, %1, off offset:16"
        :: "v"(lds), "v"(g) : "memory");
}

__device__ __forceinline__ void wait_async0() {
#if __has_builtin(__builtin_amdgcn_s_wait_asynccnt)
    __builtin_amdgcn_s_wait_asynccnt(0);
#else
    asm volatile("s_wait_asynccnt 0x0" ::: "memory");
#endif
}

// ---------------------------------------------------------------------------
// WMMA GEMM:  C[M,N] = A[Mp,Kp](bf16) @ Bt[Np,Kp](bf16, row = out column)
// Both operands zero-padded: staging is guard-free async DMA.
// Block tile 128x128, BK=32, 256 threads = 8 waves, wave tile 64x32.
// ---------------------------------------------------------------------------
#define BM 128
#define BN 128
#define BK 32

__global__ __launch_bounds__(256) void gemm_bf16_wmma(
    const __bf16* __restrict__ A, const __bf16* __restrict__ Bt,
    float* __restrict__ C, int M, int N, int Kp)
{
    __shared__ __bf16 sA[2][BM * BK];   // row-major, K contiguous
    __shared__ __bf16 sB[2][BN * BK];   // column(n)-major, K contiguous

    const int tid  = threadIdx.x;
    const int lane = tid & 31;
    const int wave = tid >> 5;
    const int mWave = (wave & 1) * 64;   // 2 waves along M
    const int nWave = (wave >> 1) * 32;  // 4 waves along N

    const int rowBase = blockIdx.y * BM;
    const int colBase = blockIdx.x * BN;

    // staging: each thread owns 32B of the A tile and 32B of the B tile
    const int sRow = tid >> 1;           // 0..127
    const int sKh  = (tid & 1) * 16;     // 0 / 16
    const __bf16* gA = A  + (size_t)(rowBase + sRow) * Kp + sKh;
    const __bf16* gB = Bt + (size_t)(colBase + sRow) * Kp + sKh;
    const unsigned lA[2] = { lds_u32(&sA[0][sRow * BK + sKh]),
                             lds_u32(&sA[1][sRow * BK + sKh]) };
    const unsigned lB[2] = { lds_u32(&sB[0][sRow * BK + sKh]),
                             lds_u32(&sB[1][sRow * BK + sKh]) };

    const int mrow = lane & 15;
    const int kbA  = (lane >> 4) * 16;   // A frag: K base 0 / 16
    const int ncol = lane & 15;
    const int kbB  = (lane >> 4) * 16;   // B frag: K base 0 / 16

    f32x8 acc[4][2];
    #pragma unroll
    for (int mi = 0; mi < 4; ++mi)
        #pragma unroll
        for (int ni = 0; ni < 2; ++ni)
            acc[mi][ni] = (f32x8)(0.0f);

    const int kTiles = Kp / BK;

    // prologue: tile 0 into buffer 0
    async_copy_32B(lA[0], gA);
    async_copy_32B(lB[0], gB);

    for (int kt = 0; kt < kTiles; ++kt) {
        const int buf = kt & 1;
        wait_async0();
        __syncthreads();

        if (kt + 1 < kTiles) {
            const __bf16* nA = gA + (size_t)(kt + 1) * BK;
            const __bf16* nB = gB + (size_t)(kt + 1) * BK;
            if (kt + 2 < kTiles) __builtin_prefetch(nA + BK, 0, 1);  // global_prefetch_b8
            async_copy_32B(lA[buf ^ 1], nA);
            async_copy_32B(lB[buf ^ 1], nB);
        }

        const __bf16* bA = sA[buf];
        const __bf16* bB = sB[buf];

        bf16x16 afrag[4];
        #pragma unroll
        for (int mi = 0; mi < 4; ++mi) {
            const __bf16* base = bA + (mWave + mi * 16 + mrow) * BK;
            afrag[mi] = *(const bf16x16*)(base + kbA);
        }
        bf16x16 bfrag[2];
        #pragma unroll
        for (int ni = 0; ni < 2; ++ni)
            bfrag[ni] = *(const bf16x16*)(bB + (nWave + ni * 16 + ncol) * BK + kbB);

        #pragma unroll
        for (int mi = 0; mi < 4; ++mi)
            #pragma unroll
            for (int ni = 0; ni < 2; ++ni)
                acc[mi][ni] = __builtin_amdgcn_wmma_f32_16x16x32_bf16(
                    false, afrag[mi], false, bfrag[ni],
                    (short)0, acc[mi][ni], false, false);

        __syncthreads();
    }

    // epilogue: D layout — VGPR r: lanes 0-15 -> M=r, lanes 16-31 -> M=r+8
    const int mHi = (lane >> 4) * 8;
    #pragma unroll
    for (int mi = 0; mi < 4; ++mi) {
        #pragma unroll
        for (int ni = 0; ni < 2; ++ni) {
            const int gn = colBase + nWave + ni * 16 + ncol;
            if (gn >= N) continue;
            #pragma unroll
            for (int r = 0; r < 8; ++r) {
                const int gm = rowBase + mWave + mi * 16 + r + mHi;
                if (gm < M) C[(size_t)gm * N + gn] = acc[mi][ni][r];
            }
        }
    }
}

// ---------------------------------------------------------------------------
// fp32 [M,K] -> zero-padded bf16 [Mp,Kp]
// ---------------------------------------------------------------------------
__global__ void cvt_h_kernel(const float* __restrict__ src, __bf16* __restrict__ dst,
                             int M, int K, int Mp, int Kp)
{
    const int kg = Kp >> 3;
    size_t i = (size_t)blockIdx.x * blockDim.x + threadIdx.x;
    if (i >= (size_t)Mp * kg) return;
    int row = (int)(i / kg);
    int kb  = (int)(i % kg) * 8;
    float v[8];
    if (row < M && kb + 8 <= K) {
        const float4* p = (const float4*)(src + (size_t)row * K + kb);
        float4 u0 = p[0], u1 = p[1];
        v[0] = u0.x; v[1] = u0.y; v[2] = u0.z; v[3] = u0.w;
        v[4] = u1.x; v[5] = u1.y; v[6] = u1.z; v[7] = u1.w;
    } else {
        #pragma unroll
        for (int j = 0; j < 8; ++j) {
            int k = kb + j;
            v[j] = (row < M && k < K) ? src[(size_t)row * K + k] : 0.0f;
        }
    }
    uint4 o;
    o.x = pack2(v[0], v[1]); o.y = pack2(v[2], v[3]);
    o.z = pack2(v[4], v[5]); o.w = pack2(v[6], v[7]);
    *(uint4*)(dst + (size_t)row * Kp + kb) = o;
}

// fp32 W[K,Nout] -> transposed zero-padded bf16 Wt[Np,Kp] (k contiguous)
__global__ void cvt_wt_kernel(const float* __restrict__ W, __bf16* __restrict__ Wt,
                              int K, int Nout, int Np, int Kp)
{
    const int kg = Kp >> 3;
    size_t i = (size_t)blockIdx.x * blockDim.x + threadIdx.x;
    if (i >= (size_t)Np * kg) return;
    int n  = (int)(i / kg);
    int kb = (int)(i % kg) * 8;
    float v[8];
    if (n < Nout && kb + 8 <= K) {
        #pragma unroll
        for (int j = 0; j < 8; ++j) v[j] = W[(size_t)(kb + j) * Nout + n];
    } else {
        #pragma unroll
        for (int j = 0; j < 8; ++j) {
            int k = kb + j;
            v[j] = (n < Nout && k < K) ? W[(size_t)k * Nout + n] : 0.0f;
        }
    }
    uint4 o;
    o.x = pack2(v[0], v[1]); o.y = pack2(v[2], v[3]);
    o.z = pack2(v[4], v[5]); o.w = pack2(v[6], v[7]);
    *(uint4*)(Wt + (size_t)n * Kp + kb) = o;
}

// ---------------------------------------------------------------------------
// h = [x[:, :768], x[:, 768:772] @ time_W + time_b]
// ---------------------------------------------------------------------------
__global__ void build_h_kernel(const float* __restrict__ x,
                               const float* __restrict__ tW,
                               const float* __restrict__ tb,
                               float* __restrict__ h, int Nn)
{
    size_t i = (size_t)blockIdx.x * blockDim.x + threadIdx.x;
    if (i >= (size_t)Nn * GEMB) return;
    int n = (int)(i / GEMB), c = (int)(i % GEMB);
    if (c < 768) {
        h[i] = x[(size_t)n * 772 + c];
    } else {
        int j = c - 768;
        float s = tb[j];
        #pragma unroll
        for (int k = 0; k < 4; ++k)
            s += x[(size_t)n * 772 + 768 + k] * tW[k * 12 + j];
        h[i] = s;
    }
}

// ---------------------------------------------------------------------------
// Attention scores: out[n,h] = sum_c hp[n, h*C+c] * att[h*C+c]   (wave32 reduce)
// ---------------------------------------------------------------------------
__global__ void att_score_kernel(const float* __restrict__ hp,
                                 const float* __restrict__ att,
                                 float* __restrict__ out,
                                 int Nn, int H, int C)
{
    int gw   = (int)(((size_t)blockIdx.x * blockDim.x + threadIdx.x) >> 5);
    int lane = threadIdx.x & 31;
    if (gw >= Nn * H) return;
    int n = gw / H, h = gw % H;
    const float* row = hp + (size_t)n * ((size_t)H * C) + (size_t)h * C;
    const float* a   = att + (size_t)h * C;
    float s = 0.0f;
    for (int c = lane; c < C; c += 32) s += row[c] * a[c];
    #pragma unroll
    for (int off = 16; off; off >>= 1) s += __shfl_down(s, off, 32);
    if (lane == 0) out[gw] = s;
}

__device__ __forceinline__ float lrelu(float x) { return x > 0.0f ? x : 0.2f * x; }

__device__ __forceinline__ void atomicMaxF(float* addr, float v) {
    if (v >= 0.0f) atomicMax((int*)addr, __float_as_int(v));
    else atomicMin((unsigned int*)addr, __float_as_uint(v));
}

__global__ void edge_max_kernel(const int* __restrict__ src, const int* __restrict__ dst,
                                const float* __restrict__ as, const float* __restrict__ ad,
                                float* __restrict__ m, int E_, int Nn, int H)
{
    int i = blockIdx.x * blockDim.x + threadIdx.x;
    int tot = (E_ + Nn) * H;
    if (i >= tot) return;
    int e = i / H, h = i % H;
    int s = e < E_ ? src[e] : e - E_;
    int d = e < E_ ? dst[e] : e - E_;
    float v = lrelu(as[s * H + h] + ad[d * H + h]);
    atomicMaxF(&m[d * H + h], v);
}

__global__ void edge_exp_kernel(const int* __restrict__ src, const int* __restrict__ dst,
                                const float* __restrict__ as, const float* __restrict__ ad,
                                const float* __restrict__ m, float* __restrict__ p,
                                float* __restrict__ denom, int E_, int Nn, int H)
{
    int i = blockIdx.x * blockDim.x + threadIdx.x;
    int tot = (E_ + Nn) * H;
    if (i >= tot) return;
    int e = i / H, h = i % H;
    int s = e < E_ ? src[e] : e - E_;
    int d = e < E_ ? dst[e] : e - E_;
    float mm = m[d * H + h];
    if (mm < -1e37f) mm = 0.0f;                 // isfinite() guard
    float v = __expf(lrelu(as[s * H + h] + ad[d * H + h]) - mm);
    p[i] = v;
    atomicAdd(&denom[d * H + h], v);
}

// one block per edge; threads stride over F = H*C columns
__global__ void edge_agg_kernel(const int* __restrict__ src, const int* __restrict__ dst,
                                const float* __restrict__ p, const float* __restrict__ denom,
                                const float* __restrict__ hp, float* __restrict__ agg,
                                int E_, int Nn, int H, int C)
{
    int e = blockIdx.x;
    int F = H * C;
    int s = e < E_ ? src[e] : e - E_;
    int d = e < E_ ? dst[e] : e - E_;
    for (int c = threadIdx.x; c < F; c += blockDim.x) {
        int h = c / C;
        float alpha = p[e * H + h] / (denom[d * H + h] + 1e-16f);
        atomicAdd(&agg[(size_t)d * F + c], alpha * hp[(size_t)s * F + c]);
    }
}

__global__ void bias_tanh_kernel(float* __restrict__ h, const float* __restrict__ bias,
                                 size_t total, int F)
{
    size_t i = (size_t)blockIdx.x * blockDim.x + threadIdx.x;
    if (i >= total) return;
    int c = (int)(i % F);
    h[i] = tanhf(h[i] + bias[c]);
}

__global__ void bn_partial_kernel(const float* __restrict__ h, float* __restrict__ sum,
                                  float* __restrict__ sumsq, int Nn, int F)
{
    int c = blockIdx.x * blockDim.x + threadIdx.x;
    if (c >= F) return;
    int chunk = blockIdx.y, nch = gridDim.y;
    int r0 = (int)(((long long)Nn * chunk) / nch);
    int r1 = (int)(((long long)Nn * (chunk + 1)) / nch);
    float s = 0.0f, q = 0.0f;
    for (int r = r0; r < r1; ++r) {
        float v = h[(size_t)r * F + c];
        s += v; q += v * v;
    }
    atomicAdd(&sum[c], s);
    atomicAdd(&sumsq[c], q);
}

__global__ void bn_apply_kernel(float* __restrict__ h, const float* __restrict__ sum,
                                const float* __restrict__ sumsq,
                                const float* __restrict__ gamma, const float* __restrict__ beta,
                                int Nn, int F)
{
    size_t i = (size_t)blockIdx.x * blockDim.x + threadIdx.x;
    if (i >= (size_t)Nn * F) return;
    int c = (int)(i % F);
    float inv = 1.0f / (float)Nn;
    float mu  = sum[c] * inv;
    float var = sumsq[c] * inv - mu * mu;
    h[i] = (h[i] - mu) * rsqrtf(var + 1e-5f) * gamma[c] + beta[c];
}

__global__ void fill_kernel(float* __restrict__ p, size_t n, float v)
{
    size_t i = (size_t)blockIdx.x * blockDim.x + threadIdx.x;
    if (i < n) p[i] = v;
}

__global__ void count_kernel(const int* __restrict__ batch, float* __restrict__ cnt, int Nn)
{
    int n = blockIdx.x * blockDim.x + threadIdx.x;
    if (n < Nn) atomicAdd(&cnt[batch[n]], 1.0f);
}

__global__ void pool_kernel(const float* __restrict__ h, const int* __restrict__ batch,
                            float* __restrict__ pooled, int Nn, int F)
{
    size_t i = (size_t)blockIdx.x * blockDim.x + threadIdx.x;
    if (i >= (size_t)Nn * F) return;
    int n = (int)(i / F), c = (int)(i % F);
    atomicAdd(&pooled[(size_t)batch[n] * F + c], h[i]);
}

__global__ void pool_finish_kernel(float* __restrict__ pooled, const float* __restrict__ cnt,
                                   int NGr, int F)
{
    int i = blockIdx.x * blockDim.x + threadIdx.x;
    if (i >= NGr * F) return;
    int g = i / F;
    float v = pooled[i] / fmaxf(cnt[g], 1.0f);
    pooled[i] = v > 0.0f ? v : 0.0f;
}

__global__ void mlp1_kernel(const float* __restrict__ pooled, const float* __restrict__ W,
                            const float* __restrict__ b, float* __restrict__ hid,
                            int NGr, int F, int Hd)
{
    int i = blockIdx.x * blockDim.x + threadIdx.x;
    if (i >= NGr * Hd) return;
    int g = i / Hd, j = i % Hd;
    float s = b[j];
    for (int c = 0; c < F; ++c) s += pooled[(size_t)g * F + c] * W[(size_t)c * Hd + j];
    hid[i] = tanhf(s);
}

__global__ void mlp2_kernel(const float* __restrict__ hid, const float* __restrict__ W,
                            const float* __restrict__ b, float* __restrict__ out,
                            int NGr, int Hd, int O)
{
    int i = blockIdx.x * blockDim.x + threadIdx.x;
    if (i >= NGr * O) return;
    int g = i / O, j = i % O;
    float s = b[j];
    for (int k = 0; k < Hd; ++k) s += hid[g * Hd + k] * W[k * O + j];
    out[i] = s;
}

// ---------------------------------------------------------------------------
// Orchestration
// ---------------------------------------------------------------------------
extern "C" void kernel_launch(void* const* d_in, const int* in_sizes, int n_in,
                              void* d_out, int out_size, void* d_ws, size_t ws_size,
                              hipStream_t stream)
{
    (void)in_sizes; (void)n_in; (void)out_size; (void)ws_size;

    const float* x      = (const float*)d_in[0];
    const int*   eidx   = (const int*)d_in[1];
    const int*   batch  = (const int*)d_in[2];
    const float* time_W = (const float*)d_in[3];
    const float* time_b = (const float*)d_in[4];

    const float* Wl[3]    = {(const float*)d_in[5],  (const float*)d_in[11], (const float*)d_in[17]};
    const float* asrc[3]  = {(const float*)d_in[6],  (const float*)d_in[12], (const float*)d_in[18]};
    const float* adst[3]  = {(const float*)d_in[7],  (const float*)d_in[13], (const float*)d_in[19]};
    const float* biasl[3] = {(const float*)d_in[8],  (const float*)d_in[14], (const float*)d_in[20]};
    const float* gam[3]   = {(const float*)d_in[9],  (const float*)d_in[15], (const float*)d_in[21]};
    const float* bet[3]   = {(const float*)d_in[10], (const float*)d_in[16], (const float*)d_in[22]};
    const float* mW1 = (const float*)d_in[23];
    const float* mb1 = (const float*)d_in[24];
    const float* mW2 = (const float*)d_in[25];
    const float* mb2 = (const float*)d_in[26];

    const int Kdim[3] = {GEMB, 3 * GEMB, 3 * GEMB};
    const int Fout[3] = {3 * GEMB, 3 * GEMB, GEMB};
    const int Hh[3]   = {3, 3, 1};
    const int KpL[3]  = {800, 2368, 2368};    // ceil(K/32)*32
    const int NpL[3]  = {2432, 2432, 896};    // ceil(F/128)*128

    const int* src = eidx;
    const int* dst = eidx + GE;

    // workspace layout (float units; all sub-offsets multiples of 4 floats)
    const size_t bigF = (size_t)GN * 3 * GEMB;      // 46.8M floats
    float* bufA   = (float*)d_ws;                   // h / agg (ping)
    float* bufB   = bufA + bigF;                    // hp
    float* a_s    = bufB + bigF;                    // N*3
    float* a_d    = a_s + (size_t)GN * 3;
    float* mbuf   = a_d + (size_t)GN * 3;
    float* denom  = mbuf + (size_t)GN * 3;
    float* pbuf   = denom + (size_t)GN * 3;         // (E+N)*3
    float* colsum = pbuf + (size_t)(GE + GN) * 3;
    float* colsq  = colsum + (size_t)3 * GEMB;
    float* pooled = colsq + (size_t)3 * GEMB;       // NG*780
    float* cnt    = pooled + (size_t)GNG * GEMB;    // NG
    float* hid    = cnt + GNG;                      // NG*128
    float* cur    = hid + (size_t)GNG * 128;
    __bf16* Ab    = (__bf16*)cur;                   // GMP x 2368 bf16
    cur += (size_t)GMP * 2368 / 2;
    __bf16* WtL[3];
    for (int L = 0; L < 3; ++L) {
        WtL[L] = (__bf16*)cur;
        cur += (size_t)NpL[L] * KpL[L] / 2;
    }

    auto cdiv = [](size_t a, size_t b) { return (unsigned)((a + b - 1) / b); };

    // pre-convert all weights: fp32 [K,F] -> bf16 transposed padded [Np,Kp]
    for (int L = 0; L < 3; ++L)
        cvt_wt_kernel<<<cdiv((size_t)NpL[L] * (KpL[L] / 8), 256), 256, 0, stream>>>(
            Wl[L], WtL[L], Kdim[L], Fout[L], NpL[L], KpL[L]);

    // h = concat(x[:, :768], time-embed)
    build_h_kernel<<<cdiv((size_t)GN * GEMB, 256), 256, 0, stream>>>(x, time_W, time_b, bufA, GN);

    float* h  = bufA;
    float* hp = bufB;

    for (int L = 0; L < 3; ++L) {
        const int K = Kdim[L], F = Fout[L], H = Hh[L], C = F / H;
        const int Kp = KpL[L], Np = NpL[L];

        // h -> padded bf16 A
        cvt_h_kernel<<<cdiv((size_t)GMP * (Kp / 8), 256), 256, 0, stream>>>(h, Ab, GN, K, GMP, Kp);

        // hp = h @ W  (bf16 WMMA, f32 accum, async-DMA staged)
        dim3 gg(Np / BN, GMP / BM);
        gemm_bf16_wmma<<<gg, 256, 0, stream>>>(Ab, WtL[L], hp, GN, F, Kp);

        // per-node attention scores
        att_score_kernel<<<cdiv((size_t)GN * H * 32, 256), 256, 0, stream>>>(hp, asrc[L], a_s, GN, H, C);
        att_score_kernel<<<cdiv((size_t)GN * H * 32, 256), 256, 0, stream>>>(hp, adst[L], a_d, GN, H, C);

        // segment softmax + aggregation (agg reuses h buffer)
        fill_kernel<<<cdiv((size_t)GN * H, 256), 256, 0, stream>>>(mbuf, (size_t)GN * H, -INFINITY);
        fill_kernel<<<cdiv((size_t)GN * H, 256), 256, 0, stream>>>(denom, (size_t)GN * H, 0.0f);
        fill_kernel<<<cdiv((size_t)GN * F, 256), 256, 0, stream>>>(h, (size_t)GN * F, 0.0f);

        const int EH = (GE + GN) * H;
        edge_max_kernel<<<cdiv(EH, 256), 256, 0, stream>>>(src, dst, a_s, a_d, mbuf, GE, GN, H);
        edge_exp_kernel<<<cdiv(EH, 256), 256, 0, stream>>>(src, dst, a_s, a_d, mbuf, pbuf, denom, GE, GN, H);
        edge_agg_kernel<<<GE + GN, 256, 0, stream>>>(src, dst, pbuf, denom, hp, h, GE, GN, H, C);

        // bias + tanh, then BatchNorm (batch statistics)
        bias_tanh_kernel<<<cdiv((size_t)GN * F, 256), 256, 0, stream>>>(h, biasl[L], (size_t)GN * F, F);
        fill_kernel<<<cdiv(F, 256), 256, 0, stream>>>(colsum, (size_t)F, 0.0f);
        fill_kernel<<<cdiv(F, 256), 256, 0, stream>>>(colsq, (size_t)F, 0.0f);
        dim3 gbn((F + 255) / 256, 64);
        bn_partial_kernel<<<gbn, 256, 0, stream>>>(h, colsum, colsq, GN, F);
        bn_apply_kernel<<<cdiv((size_t)GN * F, 256), 256, 0, stream>>>(h, colsum, colsq, gam[L], bet[L], GN, F);
        // h stays in bufA as input to next layer
    }

    // mean pool per graph + ReLU
    fill_kernel<<<cdiv((size_t)GNG * GEMB, 256), 256, 0, stream>>>(pooled, (size_t)GNG * GEMB, 0.0f);
    fill_kernel<<<1, 64, 0, stream>>>(cnt, (size_t)GNG, 0.0f);
    count_kernel<<<cdiv(GN, 256), 256, 0, stream>>>(batch, cnt, GN);
    pool_kernel<<<cdiv((size_t)GN * GEMB, 256), 256, 0, stream>>>(h, batch, pooled, GN, GEMB);
    pool_finish_kernel<<<cdiv((size_t)GNG * GEMB, 256), 256, 0, stream>>>(pooled, cnt, GNG, GEMB);

    // MLP head
    mlp1_kernel<<<cdiv(GNG * 128, 256), 256, 0, stream>>>(pooled, mW1, mb1, hid, GNG, GEMB, 128);
    mlp2_kernel<<<cdiv(GNG * 64, 64), 64, 0, stream>>>(hid, mW2, mb2, (float*)d_out, GNG, 128, 64);
}
